// LocalNodeGeometryEncoder_36936718745721
// MI455X (gfx1250) — matrix operand order repaired and checked by reading, not compiled, not run on hardware
//
#include <hip/hip_runtime.h>
#include <math.h>

// ---------------- problem constants (from the reference) ----------------
#define Bv    8
#define Nv    2048
#define BN    (Bv*Nv)        // 16384 nodes
#define RK    5              // k = min(K, N-1) = 5 neighbors
#define RHD   32             // rel token dim
#define BHD   64             // bar token dim
#define HIDv  1024
#define LNEPS 1e-5f

// ---------------- CDNA5 WMMA types ----------------
typedef __attribute__((ext_vector_type(16))) _Float16 v16h;
typedef __attribute__((ext_vector_type(8)))  float    v8f;
typedef __attribute__((ext_vector_type(4)))  float    f32x4;

union FragH { v16h v; f32x4 q[2]; };
union AccF  { v8f  v; float e[8]; };

// ---------------- small device helpers ----------------
__device__ __forceinline__ float geluf(float x) {
  return 0.5f * x * (1.0f + erff(x * 0.70710678118654752f));
}
__device__ __forceinline__ float wsum32(float v) {
  v += __shfl_xor(v, 16, 32);
  v += __shfl_xor(v,  8, 32);
  v += __shfl_xor(v,  4, 32);
  v += __shfl_xor(v,  2, 32);
  v += __shfl_xor(v,  1, 32);
  return v;
}
__device__ __forceinline__ float wsum16(float v) {
  v += __shfl_xor(v, 8, 32);
  v += __shfl_xor(v, 4, 32);
  v += __shfl_xor(v, 2, 32);
  v += __shfl_xor(v, 1, 32);
  return v;
}
__device__ __forceinline__ float wsum8(float v) {
  v += __shfl_xor(v, 4, 32);
  v += __shfl_xor(v, 2, 32);
  v += __shfl_xor(v, 1, 32);
  return v;
}

// =====================================================================
// Stage 1: top-5 of relu(adjacency row) with diagonal zeroed.
// One wave (32 lanes) per row: per-lane top-5 over a strided slice, then
// serial merge in LDS.  Adjacency (134MB) is the only HBM-bound stream:
// ~6us at 23.3 TB/s.
// =====================================================================
__global__ void __launch_bounds__(32) k_topk(const float* __restrict__ adj,
                                             float* __restrict__ topw,
                                             int*   __restrict__ topi) {
  const int row  = blockIdx.x;            // b*N + n
  const int diag = row & (Nv - 1);        // n
  const int lane = threadIdx.x;
  const float* arow = adj + (size_t)row * Nv;

  float bv[RK]; int bi[RK];
#pragma unroll
  for (int j = 0; j < RK; ++j) { bv[j] = -1.0f; bi[j] = 0x7fffffff; }

  for (int c = lane; c < Nv; c += 32) {
    if (c + 256 < Nv) __builtin_prefetch(arow + c + 256, 0, 0);  // global_prefetch_b8
    float e = arow[c];
    e = (c == diag) ? 0.0f : fmaxf(e, 0.0f);
    if (e > bv[RK-1] || (e == bv[RK-1] && c < bi[RK-1])) {
      bv[RK-1] = e; bi[RK-1] = c;
#pragma unroll
      for (int j = RK-1; j > 0; --j) {
        bool sw = (bv[j] > bv[j-1]) || (bv[j] == bv[j-1] && bi[j] < bi[j-1]);
        if (sw) {
          float tv = bv[j]; bv[j] = bv[j-1]; bv[j-1] = tv;
          int   ti = bi[j]; bi[j] = bi[j-1]; bi[j-1] = ti;
        }
      }
    }
  }

  __shared__ float sv[32 * RK];
  __shared__ int   si[32 * RK];
#pragma unroll
  for (int j = 0; j < RK; ++j) { sv[lane*RK + j] = bv[j]; si[lane*RK + j] = bi[j]; }
  __syncthreads();

  if (lane == 0) {
    for (int j = 0; j < RK; ++j) {
      float best = -2.0f; int besti = 0x7fffffff; int bestp = 0;
      for (int p = 0; p < 32 * RK; ++p) {
        float v = sv[p]; int idc = si[p];
        if (v > best || (v == best && idc < besti)) { best = v; besti = idc; bestp = p; }
      }
      sv[bestp] = -2.0f;
      topw[(size_t)row * RK + j] = best;
      topi[(size_t)row * RK + j] = (besti == 0x7fffffff) ? 0 : besti;
    }
  }
}

// =====================================================================
// Stage 2: per-node neighbor geometry (dirs, length, weight, valid mask)
// =====================================================================
__global__ void __launch_bounds__(256) k_geom(const float* __restrict__ pos,
                                              const float* __restrict__ topw,
                                              const int*   __restrict__ topi,
                                              float* __restrict__ dirs,
                                              float* __restrict__ leng,
                                              float* __restrict__ wgt,
                                              int*   __restrict__ vmask) {
  int bn = blockIdx.x * blockDim.x + threadIdx.x;
  if (bn >= BN) return;
  int b = bn / Nv;
  const float* p0 = pos + (size_t)bn * 3;
  float px = p0[0], py = p0[1], pz = p0[2];
  int mask = 0;
#pragma unroll
  for (int j = 0; j < RK; ++j) {
    float w  = topw[(size_t)bn*RK + j];
    int   ii = topi[(size_t)bn*RK + j];
    float vf = (w > 0.0f) ? 1.0f : 0.0f;
    const float* pn = pos + ((size_t)b*Nv + ii) * 3;
    float vx = pn[0]-px, vy = pn[1]-py, vz = pn[2]-pz;
    float len = sqrtf(vx*vx + vy*vy + vz*vz);
    float pl  = fmaxf(sqrtf(vx*vx + vy*vy), 1e-6f);
    dirs[((size_t)bn*RK + j)*2 + 0] = (vx / pl) * vf;
    dirs[((size_t)bn*RK + j)*2 + 1] = (vy / pl) * vf;
    leng[(size_t)bn*RK + j] = len * vf;
    wgt [(size_t)bn*RK + j] = w * vf;
    if (w > 0.0f) mask |= (1 << j);
  }
  vmask[bn] = mask;
}

// =====================================================================
// Encoder layer, d = 32 (rel).  lane == channel, wave32; weights L2-resident.
// =====================================================================
__device__ __forceinline__ void enc32(float x[RK], const bool val[RK], int lane,
    const float* __restrict__ Wqkv, const float* __restrict__ bqkv,
    const float* __restrict__ Wo,   const float* __restrict__ bo,
    const float* __restrict__ g1,   const float* __restrict__ be1,
    const float* __restrict__ g2,   const float* __restrict__ be2,
    const float* __restrict__ Wf1,  const float* __restrict__ bf1,
    const float* __restrict__ Wf2,  const float* __restrict__ bf2) {
  const float ln1g = g1[lane], ln1b = be1[lane];
  const float ln2g = g2[lane], ln2b = be2[lane];
  float q[RK], kk[RK], vv[RK];
#pragma unroll
  for (int j = 0; j < RK; ++j) {
    float mu  = wsum32(x[j]) * (1.0f/32.0f);
    float d   = x[j] - mu;
    float var = wsum32(d*d) * (1.0f/32.0f);
    float h   = d * rsqrtf(var + LNEPS) * ln1g + ln1b;
    float aq = bqkv[lane], ak = bqkv[lane+32], av = bqkv[lane+64];
    for (int c = 0; c < 32; ++c) {
      float hc = __shfl(h, c, 32);
      aq = fmaf(hc, Wqkv[(lane     )*32 + c], aq);
      ak = fmaf(hc, Wqkv[(lane + 32)*32 + c], ak);
      av = fmaf(hc, Wqkv[(lane + 64)*32 + c], av);
    }
    q[j] = aq; kk[j] = ak; vv[j] = av;
  }
  const float scale = 0.35355339059327376f;  // 1/sqrt(8), dh=8 (NH=4)
  float o[RK];
#pragma unroll
  for (int s = 0; s < RK; ++s) {
    float sc[RK];
#pragma unroll
    for (int t = 0; t < RK; ++t) {
      float p = wsum8(q[s] * kk[t]) * scale;   // head = lane>>3 (8-lane groups)
      sc[t] = val[t] ? p : -1e30f;
    }
    float mx = sc[0];
#pragma unroll
    for (int t = 1; t < RK; ++t) mx = fmaxf(mx, sc[t]);
    float se = 0.0f;
#pragma unroll
    for (int t = 0; t < RK; ++t) { sc[t] = __expf(sc[t] - mx); se += sc[t]; }
    float inv = 1.0f / se, acc = 0.0f;
#pragma unroll
    for (int t = 0; t < RK; ++t) acc = fmaf(sc[t] * inv, vv[t], acc);
    o[s] = acc;
  }
#pragma unroll
  for (int j = 0; j < RK; ++j) {
    float a = bo[lane];
    for (int c = 0; c < 32; ++c) a = fmaf(__shfl(o[j], c, 32), Wo[lane*32 + c], a);
    x[j] += a;
  }
#pragma unroll
  for (int j = 0; j < RK; ++j) {
    float mu  = wsum32(x[j]) * (1.0f/32.0f);
    float d   = x[j] - mu;
    float var = wsum32(d*d) * (1.0f/32.0f);
    float h   = d * rsqrtf(var + LNEPS) * ln2g + ln2b;
    float f0 = bf1[lane], f1 = bf1[lane+32], f2 = bf1[lane+64], f3 = bf1[lane+96];
    for (int c = 0; c < 32; ++c) {
      float hc = __shfl(h, c, 32);
      f0 = fmaf(hc, Wf1[(lane     )*32 + c], f0);
      f1 = fmaf(hc, Wf1[(lane + 32)*32 + c], f1);
      f2 = fmaf(hc, Wf1[(lane + 64)*32 + c], f2);
      f3 = fmaf(hc, Wf1[(lane + 96)*32 + c], f3);
    }
    f0 = geluf(f0); f1 = geluf(f1); f2 = geluf(f2); f3 = geluf(f3);
    float a = bf2[lane];
    for (int c = 0; c < 32; ++c) {
      a = fmaf(__shfl(f0, c, 32), Wf2[lane*128 +      c], a);
      a = fmaf(__shfl(f1, c, 32), Wf2[lane*128 + 32 + c], a);
      a = fmaf(__shfl(f2, c, 32), Wf2[lane*128 + 64 + c], a);
      a = fmaf(__shfl(f3, c, 32), Wf2[lane*128 + 96 + c], a);
    }
    x[j] += a;
  }
}

// =====================================================================
// Encoder layer, d = 64 (bar).  lane holds channels c0=lane, c1=lane+32.
// NH=4, dh=16 -> head(c0)=lane>>4 in {0,1}, head(c1)=2+(lane>>4).
// =====================================================================
__device__ __forceinline__ void enc64(float x0[RK], float x1[RK], const bool val[RK], int lane,
    const float* __restrict__ Wqkv, const float* __restrict__ bqkv,
    const float* __restrict__ Wo,   const float* __restrict__ bo,
    const float* __restrict__ g1,   const float* __restrict__ be1,
    const float* __restrict__ g2,   const float* __restrict__ be2,
    const float* __restrict__ Wf1,  const float* __restrict__ bf1,
    const float* __restrict__ Wf2,  const float* __restrict__ bf2) {
  const int c0 = lane, c1 = lane + 32;
  float q0[RK], q1[RK], k0[RK], k1[RK], v0[RK], v1[RK];
#pragma unroll
  for (int j = 0; j < RK; ++j) {
    float mu  = wsum32(x0[j] + x1[j]) * (1.0f/64.0f);
    float d0  = x0[j] - mu, d1 = x1[j] - mu;
    float var = wsum32(d0*d0 + d1*d1) * (1.0f/64.0f);
    float rs  = rsqrtf(var + LNEPS);
    float h0  = d0 * rs * g1[c0] + be1[c0];
    float h1  = d1 * rs * g1[c1] + be1[c1];
    float aq0 = bqkv[c0],      aq1 = bqkv[c1];
    float ak0 = bqkv[64 + c0], ak1 = bqkv[64 + c1];
    float av0 = bqkv[128 + c0], av1 = bqkv[128 + c1];
    for (int c = 0; c < 32; ++c) {
      float hc = __shfl(h0, c, 32);
      aq0 = fmaf(hc, Wqkv[(c0      )*64 + c], aq0);
      aq1 = fmaf(hc, Wqkv[(c1      )*64 + c], aq1);
      ak0 = fmaf(hc, Wqkv[(64  + c0)*64 + c], ak0);
      ak1 = fmaf(hc, Wqkv[(64  + c1)*64 + c], ak1);
      av0 = fmaf(hc, Wqkv[(128 + c0)*64 + c], av0);
      av1 = fmaf(hc, Wqkv[(128 + c1)*64 + c], av1);
    }
    for (int c = 0; c < 32; ++c) {
      float hc = __shfl(h1, c, 32);
      aq0 = fmaf(hc, Wqkv[(c0      )*64 + 32 + c], aq0);
      aq1 = fmaf(hc, Wqkv[(c1      )*64 + 32 + c], aq1);
      ak0 = fmaf(hc, Wqkv[(64  + c0)*64 + 32 + c], ak0);
      ak1 = fmaf(hc, Wqkv[(64  + c1)*64 + 32 + c], ak1);
      av0 = fmaf(hc, Wqkv[(128 + c0)*64 + 32 + c], av0);
      av1 = fmaf(hc, Wqkv[(128 + c1)*64 + 32 + c], av1);
    }
    q0[j]=aq0; q1[j]=aq1; k0[j]=ak0; k1[j]=ak1; v0[j]=av0; v1[j]=av1;
  }
  const float scale = 0.25f;  // 1/sqrt(16)
  float o0[RK], o1[RK];
#pragma unroll
  for (int s = 0; s < RK; ++s) {
    float s0[RK], s1[RK];
#pragma unroll
    for (int t = 0; t < RK; ++t) {
      float p0 = wsum16(q0[s] * k0[t]) * scale;  // 16-lane groups = heads 0/1
      float p1 = wsum16(q1[s] * k1[t]) * scale;  // heads 2/3
      s0[t] = val[t] ? p0 : -1e30f;
      s1[t] = val[t] ? p1 : -1e30f;
    }
    float m0 = s0[0], m1 = s1[0];
#pragma unroll
    for (int t = 1; t < RK; ++t) { m0 = fmaxf(m0, s0[t]); m1 = fmaxf(m1, s1[t]); }
    float e0 = 0.f, e1 = 0.f;
#pragma unroll
    for (int t = 0; t < RK; ++t) {
      s0[t] = __expf(s0[t] - m0); e0 += s0[t];
      s1[t] = __expf(s1[t] - m1); e1 += s1[t];
    }
    float i0 = 1.0f / e0, i1 = 1.0f / e1, a0 = 0.f, a1 = 0.f;
#pragma unroll
    for (int t = 0; t < RK; ++t) {
      a0 = fmaf(s0[t] * i0, v0[t], a0);
      a1 = fmaf(s1[t] * i1, v1[t], a1);
    }
    o0[s] = a0; o1[s] = a1;
  }
#pragma unroll
  for (int j = 0; j < RK; ++j) {
    float a0 = bo[c0], a1 = bo[c1];
    for (int c = 0; c < 32; ++c) {
      float oc = __shfl(o0[j], c, 32);
      a0 = fmaf(oc, Wo[c0*64 + c], a0);
      a1 = fmaf(oc, Wo[c1*64 + c], a1);
    }
    for (int c = 0; c < 32; ++c) {
      float oc = __shfl(o1[j], c, 32);
      a0 = fmaf(oc, Wo[c0*64 + 32 + c], a0);
      a1 = fmaf(oc, Wo[c1*64 + 32 + c], a1);
    }
    x0[j] += a0; x1[j] += a1;
  }
#pragma unroll
  for (int j = 0; j < RK; ++j) {
    float mu  = wsum32(x0[j] + x1[j]) * (1.0f/64.0f);
    float d0  = x0[j] - mu, d1 = x1[j] - mu;
    float var = wsum32(d0*d0 + d1*d1) * (1.0f/64.0f);
    float rs  = rsqrtf(var + LNEPS);
    float h0  = d0 * rs * g2[c0] + be2[c0];
    float h1  = d1 * rs * g2[c1] + be2[c1];
    float f[8];
#pragma unroll
    for (int g = 0; g < 8; ++g) f[g] = bf1[lane + 32*g];
    for (int c = 0; c < 32; ++c) {
      float hc = __shfl(h0, c, 32);
#pragma unroll
      for (int g = 0; g < 8; ++g) f[g] = fmaf(hc, Wf1[(lane + 32*g)*64 + c], f[g]);
    }
    for (int c = 0; c < 32; ++c) {
      float hc = __shfl(h1, c, 32);
#pragma unroll
      for (int g = 0; g < 8; ++g) f[g] = fmaf(hc, Wf1[(lane + 32*g)*64 + 32 + c], f[g]);
    }
#pragma unroll
    for (int g = 0; g < 8; ++g) f[g] = geluf(f[g]);
    float a0 = bf2[c0], a1 = bf2[c1];
#pragma unroll
    for (int g = 0; g < 8; ++g) {
      for (int c = 0; c < 32; ++c) {
        float fv = __shfl(f[g], c, 32);
        a0 = fmaf(fv, Wf2[c0*256 + 32*g + c], a0);
        a1 = fmaf(fv, Wf2[c1*256 + 32*g + c], a1);
      }
    }
    x0[j] += a0; x1[j] += a1;
  }
}

// =====================================================================
// Stage 3+4: wave per (b,n,i) set: pair features -> pair MLP -> enc32 ->
// masked mean -> rel_sum[bn][i][32]
// =====================================================================
__global__ void __launch_bounds__(256) k_rel(
    const float* __restrict__ dirs, const float* __restrict__ leng,
    const float* __restrict__ wgt,  const int* __restrict__ vmask,
    const float* __restrict__ Wp1, const float* __restrict__ bp1,
    const float* __restrict__ Wp2, const float* __restrict__ bp2,
    const float* __restrict__ Wqkv, const float* __restrict__ bqkv,
    const float* __restrict__ Wo,   const float* __restrict__ bo,
    const float* __restrict__ g1,   const float* __restrict__ be1,
    const float* __restrict__ g2,   const float* __restrict__ be2,
    const float* __restrict__ Wf1,  const float* __restrict__ bf1,
    const float* __restrict__ Wf2,  const float* __restrict__ bf2,
    float* __restrict__ rel_sum) {
  const int lane = threadIdx.x & 31;
  const int wid  = threadIdx.x >> 5;
  const int s    = blockIdx.x * 8 + wid;
  if (s >= BN * RK) return;
  const int bn = s / RK;
  const int i  = s % RK;

  float dx[RK], dy[RK], ln[RK], ww[RK];
  const int m = vmask[bn];
#pragma unroll
  for (int j = 0; j < RK; ++j) {
    dx[j] = dirs[((size_t)bn*RK + j)*2 + 0];
    dy[j] = dirs[((size_t)bn*RK + j)*2 + 1];
    ln[j] = leng[(size_t)bn*RK + j];
    ww[j] = wgt [(size_t)bn*RK + j];
  }
  bool val[RK];
#pragma unroll
  for (int j = 0; j < RK; ++j)
    val[j] = ((m >> i) & 1) && ((m >> j) & 1) && (j != i);
  bool empty = !(val[0] | val[1] | val[2] | val[3] | val[4]);

  float x[RK];
#pragma unroll
  for (int j = 0; j < RK; ++j) {
    float dot = fminf(fmaxf(dx[i]*dx[j] + dy[i]*dy[j], -1.0f), 1.0f);
    float sn  = fminf(fabsf(dx[i]*dy[j] - dy[i]*dx[j]), 1.0f);
    float f[6] = { dot, sn, ln[i], ln[j], ww[i], ww[j] };
    float t = bp1[lane];
#pragma unroll
    for (int ff = 0; ff < 6; ++ff) t = fmaf(f[ff], Wp1[lane*6 + ff], t);
    t = geluf(t);
    float acc = bp2[lane];
    for (int c = 0; c < 32; ++c)
      acc = fmaf(__shfl(t, c, 32), Wp2[lane*32 + c], acc);
    x[j] = acc;
  }
  if (empty) { x[0] = 0.0f; val[0] = true; }  // _encode_set empty-set fix

  enc32(x, val, lane, Wqkv, bqkv, Wo, bo, g1, be1, g2, be2, Wf1, bf1, Wf2, bf2);

  float num = 0.0f, den = 0.0f;
#pragma unroll
  for (int j = 0; j < RK; ++j) {
    float wv = val[j] ? 1.0f : 0.0f;
    num = fmaf(x[j], wv, num);
    den += wv;
  }
  rel_sum[((size_t)bn*RK + i)*RHD + lane] = num / fmaxf(den, 1.0f);
}

// =====================================================================
// Stage 5+6: wave per (b,n): barf MLP + rel_sum + up MLP -> enc64 ->
// masked mean -> node (f16, ready for WMMA)
// =====================================================================
__global__ void __launch_bounds__(256) k_bar(
    const float* __restrict__ dirs, const float* __restrict__ leng,
    const float* __restrict__ wgt,  const int* __restrict__ vmask,
    const float* __restrict__ rel_sum,
    const float* __restrict__ Wb1, const float* __restrict__ bb1,
    const float* __restrict__ Wb2, const float* __restrict__ bb2,
    const float* __restrict__ Wu1, const float* __restrict__ bu1,
    const float* __restrict__ Wu2, const float* __restrict__ bu2,
    const float* __restrict__ Wqkv, const float* __restrict__ bqkv,
    const float* __restrict__ Wo,   const float* __restrict__ bo,
    const float* __restrict__ g1,   const float* __restrict__ be1,
    const float* __restrict__ g2,   const float* __restrict__ be2,
    const float* __restrict__ Wf1,  const float* __restrict__ bf1,
    const float* __restrict__ Wf2,  const float* __restrict__ bf2,
    _Float16* __restrict__ node_h) {
  const int lane = threadIdx.x & 31;
  const int wid  = threadIdx.x >> 5;
  const int bn   = blockIdx.x * 8 + wid;
  if (bn >= BN) return;
  const int m = vmask[bn];
  bool val[RK];
#pragma unroll
  for (int j = 0; j < RK; ++j) val[j] = (m >> j) & 1;
  const bool empty = (m == 0);

  float x0[RK], x1[RK];
#pragma unroll
  for (int j = 0; j < RK; ++j) {
    float f[4] = { dirs[((size_t)bn*RK + j)*2 + 0],
                   dirs[((size_t)bn*RK + j)*2 + 1],
                   leng[(size_t)bn*RK + j],
                   wgt [(size_t)bn*RK + j] };
    float t = bb1[lane];
#pragma unroll
    for (int ff = 0; ff < 4; ++ff) t = fmaf(f[ff], Wb1[lane*4 + ff], t);
    t = geluf(t);
    float t2 = bb2[lane];
    for (int c = 0; c < 32; ++c)
      t2 = fmaf(__shfl(t, c, 32), Wb2[lane*32 + c], t2);
    t2 += rel_sum[((size_t)bn*RK + j)*RHD + lane];   // bar_tok (32-dim)
    // up1: 32 -> 64, gelu
    float u0 = bu1[lane], u1 = bu1[lane + 32];
    for (int c = 0; c < 32; ++c) {
      float tc = __shfl(t2, c, 32);
      u0 = fmaf(tc, Wu1[(lane     )*32 + c], u0);
      u1 = fmaf(tc, Wu1[(lane + 32)*32 + c], u1);
    }
    u0 = geluf(u0); u1 = geluf(u1);
    // up2: 64 -> 64
    float a0 = bu2[lane], a1 = bu2[lane + 32];
    for (int c = 0; c < 32; ++c) {
      float uc = __shfl(u0, c, 32);
      a0 = fmaf(uc, Wu2[(lane     )*64 + c], a0);
      a1 = fmaf(uc, Wu2[(lane + 32)*64 + c], a1);
    }
    for (int c = 0; c < 32; ++c) {
      float uc = __shfl(u1, c, 32);
      a0 = fmaf(uc, Wu2[(lane     )*64 + 32 + c], a0);
      a1 = fmaf(uc, Wu2[(lane + 32)*64 + 32 + c], a1);
    }
    float vf = val[j] ? 1.0f : 0.0f;
    x0[j] = a0 * vf; x1[j] = a1 * vf;
  }
  if (empty) { x0[0] = 0.0f; x1[0] = 0.0f; val[0] = true; }

  enc64(x0, x1, val, lane, Wqkv, bqkv, Wo, bo, g1, be1, g2, be2, Wf1, bf1, Wf2, bf2);

  float n0 = 0.f, n1 = 0.f, den = 0.f;
#pragma unroll
  for (int j = 0; j < RK; ++j) {
    float wv = val[j] ? 1.0f : 0.0f;
    n0 = fmaf(x0[j], wv, n0);
    n1 = fmaf(x1[j], wv, n1);
    den += wv;
  }
  float inv = 1.0f / fmaxf(den, 1.0f);
  node_h[(size_t)bn*BHD + lane]      = (_Float16)(n0 * inv);
  node_h[(size_t)bn*BHD + lane + 32] = (_Float16)(n1 * inv);
}

// =====================================================================
// f32 -> f16 weight prep
// =====================================================================
__global__ void __launch_bounds__(256) k_f32h(const float* __restrict__ src,
                                              _Float16* __restrict__ dst, int n) {
  int i = blockIdx.x * blockDim.x + threadIdx.x;
  if (i < n) dst[i] = (_Float16)src[i];
}

// =====================================================================
// WMMA GEMM fragment loaders (CDNA5 v_wmma_f32_16x16x32_f16 layouts).
// A (16x32 f16): lane half=0: regs0-3=K[0..7], regs4-7=K[16..23];
//                half=1: K[8..15], K[24..31].  Two 16B loads per frag.
// B (32x16 f16): same K order as A (half 0: K[0..7],K[16..23]; half 1: K[8..15],K[24..31]).
// Both A rows (M x K) and weight rows (N x K) are row-major along K, so
// OUT[m,n] = sum_k A[m,k] * W[n,k] maps directly onto D = A x B.
// Each wave computes a 16x64 strip (4 N-tiles): one A fragment feeds
// 4 v_wmma per K-step, 1.6x the FLOP:byte of the 1-tile version and 4x
// fewer waves re-reading W from L2.
// =====================================================================
#define NTW 4   // N-tiles per wave

__device__ __forceinline__ v16h load_fragA(const _Float16* __restrict__ A, int ldk,
                                           int m0, int kk, int lane) {
  const int mrow = m0 + (lane & 15);
  const int half = lane >> 4;
  const _Float16* p = A + (size_t)mrow * ldk + kk;
  FragH f;
  f.q[0] = *(const f32x4*)(p + half * 8);        // regs 0-3
  f.q[1] = *(const f32x4*)(p + 16 + half * 8);   // regs 4-7
  return f.v;
}
__device__ __forceinline__ v16h load_fragB(const _Float16* __restrict__ W, int ldk,
                                           int n0, int kk, int lane) {
  const int nrow = n0 + (lane & 15);
  const int half = lane >> 4;
  const _Float16* p = W + (size_t)nrow * ldk + kk;
  FragH f;
  f.q[0] = *(const f32x4*)(p + half * 8);        // regs 0-3: K = half*8 .. +7
  f.q[1] = *(const f32x4*)(p + 16 + half * 8);   // regs 4-7: K = 16+half*8 .. +7
  return f.v;
}

// H = gelu(node @ W1^T + b1), M=16384, K=64, N=1024 -> f16
__global__ void __launch_bounds__(256) k_out1(const _Float16* __restrict__ node,
                                              const _Float16* __restrict__ W1,
                                              const float* __restrict__ b1,
                                              _Float16* __restrict__ H) {
  const int lane = threadIdx.x & 31;
  const int wid  = threadIdx.x >> 5;
  const int grp  = blockIdx.x * 8 + wid;         // 16x64 strip index
  const int NG   = HIDv / (16 * NTW);            // 16 strips along N
  const int m0   = (grp / NG) * 16;
  const int n0   = (grp % NG) * (16 * NTW);
  AccF acc[NTW];
#pragma unroll
  for (int t = 0; t < NTW; ++t) {
    float bias = b1[n0 + 16*t + (lane & 15)];    // C init = bias (N dep. only)
#pragma unroll
    for (int r = 0; r < 8; ++r) acc[t].e[r] = bias;
  }
#pragma unroll
  for (int kk = 0; kk < BHD; kk += 32) {
    v16h a = load_fragA(node, BHD, m0, kk, lane);
#pragma unroll
    for (int t = 0; t < NTW; ++t) {
      v16h b = load_fragB(W1, BHD, n0 + 16*t, kk, lane);
      acc[t].v = __builtin_amdgcn_wmma_f32_16x16x32_f16(false, a, false, b,
                                                        (short)0, acc[t].v, false, false);
    }
  }
  const int half = lane >> 4;
#pragma unroll
  for (int t = 0; t < NTW; ++t) {
    const int nc = n0 + 16*t + (lane & 15);
#pragma unroll
    for (int r = 0; r < 8; ++r) {
      int mr = m0 + r + half * 8;
      H[(size_t)mr * HIDv + nc] = (_Float16)geluf(acc[t].e[r]);
    }
  }
}

// OUT = H @ W2^T + b2, M=16384, K=1024, N=1024 -> f32 (final output)
__global__ void __launch_bounds__(256) k_out2(const _Float16* __restrict__ H,
                                              const _Float16* __restrict__ W2,
                                              const float* __restrict__ b2,
                                              float* __restrict__ out) {
  const int lane = threadIdx.x & 31;
  const int wid  = threadIdx.x >> 5;
  const int grp  = blockIdx.x * 8 + wid;
  const int NG   = HIDv / (16 * NTW);
  const int m0   = (grp / NG) * 16;
  const int n0   = (grp % NG) * (16 * NTW);
  AccF acc[NTW];
#pragma unroll
  for (int t = 0; t < NTW; ++t) {
    float bias = b2[n0 + 16*t + (lane & 15)];
#pragma unroll
    for (int r = 0; r < 8; ++r) acc[t].e[r] = bias;
  }
  for (int kk = 0; kk < HIDv; kk += 32) {
    v16h a = load_fragA(H, HIDv, m0, kk, lane);
#pragma unroll
    for (int t = 0; t < NTW; ++t) {
      v16h b = load_fragB(W2, HIDv, n0 + 16*t, kk, lane);
      acc[t].v = __builtin_amdgcn_wmma_f32_16x16x32_f16(false, a, false, b,
                                                        (short)0, acc[t].v, false, false);
    }
  }
  const int half = lane >> 4;
#pragma unroll
  for (int t = 0; t < NTW; ++t) {
    const int nc = n0 + 16*t + (lane & 15);
#pragma unroll
    for (int r = 0; r < 8; ++r) {
      int mr = m0 + r + half * 8;
      out[(size_t)mr * HIDv + nc] = acc[t].e[r];
    }
  }
}

// =====================================================================
// Host launcher.  d_in flattening assumption: setup_inputs() insertion
// order, params dict in reference-file insertion order, each linear as
// {w, b}, each encoder layer as {in_proj_w, in_proj_b, out_proj{w,b},
// ln1_g, ln1_b, ln2_g, ln2_b, ff1{w,b}, ff2{w,b}}.
// =====================================================================
extern "C" void kernel_launch(void* const* d_in, const int* in_sizes, int n_in,
                              void* d_out, int out_size, void* d_ws, size_t ws_size,
                              hipStream_t stream) {
  (void)in_sizes; (void)n_in; (void)out_size; (void)ws_size;

  const float* positions = (const float*)d_in[0];
  const float* adjacency = (const float*)d_in[1];
  const float* pair1_w = (const float*)d_in[2];
  const float* pair1_b = (const float*)d_in[3];
  const float* pair2_w = (const float*)d_in[4];
  const float* pair2_b = (const float*)d_in[5];
  const float* barf1_w = (const float*)d_in[6];
  const float* barf1_b = (const float*)d_in[7];
  const float* barf2_w = (const float*)d_in[8];
  const float* barf2_b = (const float*)d_in[9];
  const float* up1_w = (const float*)d_in[10];
  const float* up1_b = (const float*)d_in[11];
  const float* up2_w = (const float*)d_in[12];
  const float* up2_b = (const float*)d_in[13];
  const float* rel_qkv_w = (const float*)d_in[14];
  const float* rel_qkv_b = (const float*)d_in[15];
  const float* rel_out_w = (const float*)d_in[16];
  const float* rel_out_b = (const float*)d_in[17];
  const float* rel_ln1_g = (const float*)d_in[18];
  const float* rel_ln1_b = (const float*)d_in[19];
  const float* rel_ln2_g = (const float*)d_in[20];
  const float* rel_ln2_b = (const float*)d_in[21];
  const float* rel_ff1_w = (const float*)d_in[22];
  const float* rel_ff1_b = (const float*)d_in[23];
  const float* rel_ff2_w = (const float*)d_in[24];
  const float* rel_ff2_b = (const float*)d_in[25];
  const float* bar_qkv_w = (const float*)d_in[26];
  const float* bar_qkv_b = (const float*)d_in[27];
  const float* bar_out_w = (const float*)d_in[28];
  const float* bar_out_b = (const float*)d_in[29];
  const float* bar_ln1_g = (const float*)d_in[30];
  const float* bar_ln1_b = (const float*)d_in[31];
  const float* bar_ln2_g = (const float*)d_in[32];
  const float* bar_ln2_b = (const float*)d_in[33];
  const float* bar_ff1_w = (const float*)d_in[34];
  const float* bar_ff1_b = (const float*)d_in[35];
  const float* bar_ff2_w = (const float*)d_in[36];
  const float* bar_ff2_b = (const float*)d_in[37];
  const float* out1_w = (const float*)d_in[38];
  const float* out1_b = (const float*)d_in[39];
  const float* out2_w = (const float*)d_in[40];
  const float* out2_b = (const float*)d_in[41];

  // ---- workspace bump allocator (~50 MB) ----
  size_t off = 0;
  auto wsalloc = [&](size_t bytes) -> void* {
    off = (off + 255) & ~(size_t)255;
    void* p = (char*)d_ws + off;
    off += bytes;
    return p;
  };
  float*    topw    = (float*)   wsalloc((size_t)BN * RK * 4);
  int*      topi    = (int*)     wsalloc((size_t)BN * RK * 4);
  float*    dirsb   = (float*)   wsalloc((size_t)BN * RK * 2 * 4);
  float*    lengb   = (float*)   wsalloc((size_t)BN * RK * 4);
  float*    wgtb    = (float*)   wsalloc((size_t)BN * RK * 4);
  int*      vmaskb  = (int*)     wsalloc((size_t)BN * 4);
  float*    rel_sum = (float*)   wsalloc((size_t)BN * RK * RHD * 4);
  _Float16* node_h  = (_Float16*)wsalloc((size_t)BN * BHD * 2);
  _Float16* W1h     = (_Float16*)wsalloc((size_t)HIDv * BHD * 2);
  _Float16* W2h     = (_Float16*)wsalloc((size_t)HIDv * HIDv * 2);
  _Float16* Hh      = (_Float16*)wsalloc((size_t)BN * HIDv * 2);

  // 1) top-k (HBM-bound: 134 MB adjacency stream)
  k_topk<<<BN, 32, 0, stream>>>(adjacency, topw, topi);
  // 2) neighbor geometry
  k_geom<<<BN / 256, 256, 0, stream>>>(positions, topw, topi, dirsb, lengb, wgtb, vmaskb);
  // weight f32->f16 prep (overlaps with stages 3-5)
  k_f32h<<<(HIDv * BHD + 255) / 256, 256, 0, stream>>>(out1_w, W1h, HIDv * BHD);
  k_f32h<<<(HIDv * HIDv + 255) / 256, 256, 0, stream>>>(out2_w, W2h, HIDv * HIDv);
  // 3+4) rel set encoder (81920 waves)
  k_rel<<<BN * RK / 8, 256, 0, stream>>>(
      dirsb, lengb, wgtb, vmaskb,
      pair1_w, pair1_b, pair2_w, pair2_b,
      rel_qkv_w, rel_qkv_b, rel_out_w, rel_out_b,
      rel_ln1_g, rel_ln1_b, rel_ln2_g, rel_ln2_b,
      rel_ff1_w, rel_ff1_b, rel_ff2_w, rel_ff2_b,
      rel_sum);
  // 5+6) bar set encoder -> node (f16)
  k_bar<<<BN / 8, 256, 0, stream>>>(
      dirsb, lengb, wgtb, vmaskb, rel_sum,
      barf1_w, barf1_b, barf2_w, barf2_b,
      up1_w, up1_b, up2_w, up2_b,
      bar_qkv_w, bar_qkv_b, bar_out_w, bar_out_b,
      bar_ln1_g, bar_ln1_b, bar_ln2_g, bar_ln2_b,
      bar_ff1_w, bar_ff1_b, bar_ff2_w, bar_ff2_b,
      node_h);
  // 7) output MLP via WMMA (f16 in, f32 accumulate), 16x64 strip per wave
  k_out1<<<(BN / 16) * (HIDv / (16 * NTW)) / 8, 256, 0, stream>>>(node_h, W1h, out1_b, Hh);
  k_out2<<<(BN / 16) * (HIDv / (16 * NTW)) / 8, 256, 0, stream>>>(Hh, W2h, out2_b, (float*)d_out);
}
